// GC_Transformer_54589034332455
// MI455X (gfx1250) — matrix operand order repaired and checked by reading, not compiled, not run on hardware
//
#include <hip/hip_runtime.h>
#include <hip/hip_bf16.h>

// ---------------------------------------------------------------------------
// GC_Transformer for MI455X (gfx1250): WMMA f16 GEMMs + fused epilogues.
// B=32,T=24,C=512 -> N=16384 seqs, M=393216 rows, HID=36, FFN=144, 2 layers.
// ~21 GFLOP vs ~350MB traffic at 23.3 TB/s: bandwidth-bound. GEMMs use
// v_wmma_f32_16x16x32_f16 (f32 accumulate); residual tiles are DMA'd to LDS
// with global_load_async_to_lds_b128 overlapping the WMMA loop; epilogues go
// through an LDS tile so all global stores are fully coalesced.
// ---------------------------------------------------------------------------

typedef __attribute__((ext_vector_type(16))) _Float16 v16h;
typedef __attribute__((ext_vector_type(8)))  _Float16 v8h;
typedef __attribute__((ext_vector_type(8)))  float    v8f;

#define GB  32
#define GT  24
#define GC  512
#define GIN 14
#define GHID 36
#define GFFN 144
#define GHEADS 2
#define GHD 18
#define NSEQ (GB * GC)              // 16384
#define MROWS (NSEQ * GT)           // 393216  (multiple of 64)

// gfx1250 async Global->LDS copy (16B per lane), tracked by ASYNCcnt.
__device__ __forceinline__ void async_g2l_b128(void* lds_dst, const float* gsrc) {
  unsigned ldsa = (unsigned)(unsigned long long)(uintptr_t)lds_dst;
  asm volatile("global_load_async_to_lds_b128 %0, %1, off"
               :: "v"(ldsa), "v"((unsigned long long)(uintptr_t)gsrc)
               : "memory");
}
__device__ __forceinline__ void wait_asynccnt0() {
  asm volatile("s_wait_asynccnt 0" ::: "memory");
}

// ---------------------------------------------------------------------------
// utility kernels
// ---------------------------------------------------------------------------
__global__ void zero_f32(float* __restrict__ p, long n) {
  long i = (long)blockIdx.x * blockDim.x + threadIdx.x;
  long stride = (long)gridDim.x * blockDim.x;
  for (; i < n; i += stride) p[i] = 0.0f;
}

// xn[n,t,14] = concat(pm25[b,t,c], feature[b,t,c,:13]),  n = b*C + c
__global__ void build_xn(const float* __restrict__ pm25,
                         const float* __restrict__ feat,
                         float* __restrict__ xn) {
  int idx = blockIdx.x * blockDim.x + threadIdx.x;   // n*T + t
  if (idx >= NSEQ * GT) return;
  int n = idx / GT, t = idx % GT;
  int b = n / GC,  c = n % GC;
  float* o = xn + (long)idx * GIN;
  o[0] = pm25[((long)b * GT + t) * GC + c];
  const float* f = feat + (((long)b * (2 * GT) + t) * GC + c) * (GIN - 1);
  #pragma unroll
  for (int i = 0; i < GIN - 1; i++) o[1 + i] = f[i];
}

__global__ void deg_kernel(const int* __restrict__ edge, int EG,
                           float* __restrict__ deg) {
  int idx = blockIdx.x * blockDim.x + threadIdx.x;   // b*EG + e
  if (idx >= GB * EG) return;
  int b = idx / EG, e = idx % EG;
  atomicAdd(&deg[edge[e] + b * GC], 1.0f);
}

__global__ void dinv_kernel(const float* __restrict__ deg,
                            float* __restrict__ dinv) {
  int i = blockIdx.x * blockDim.x + threadIdx.x;
  if (i >= NSEQ) return;
  float d = deg[i];
  dinv[i] = (d > 0.0f) ? rsqrtf(fmaxf(d, 1.0f)) : 0.0f;
}

// tx1[col] += xn[row] * (-(dinv[row]*dinv[col]))  per (edge, t)
__global__ void scatter_kernel(const int* __restrict__ edge, int EG,
                               const float* __restrict__ dinv,
                               const float* __restrict__ xn,
                               float* __restrict__ tx1) {
  int idx = blockIdx.x * blockDim.x + threadIdx.x;   // (b*EG + e)*T + t
  if (idx >= GB * EG * GT) return;
  int t  = idx % GT;
  int be = idx / GT;
  int b  = be / EG, e = be % EG;
  int rn = edge[e] + b * GC;
  int cn = edge[EG + e] + b * GC;
  float w = -(dinv[rn] * dinv[cn]);
  const float* src = xn  + ((long)rn * GT + t) * GIN;
  float*       dst = tx1 + ((long)cn * GT + t) * GIN;
  #pragma unroll
  for (int f = 0; f < GIN; f++) atomicAdd(&dst[f], src[f] * w);
}

// g = sigmoid(xn@W0 + tx1@W1 + b); h = concat(xn2, g)@tin_W + tin_b + PE(t)
__global__ void fuse_h_kernel(const float* __restrict__ pm25,
                              const float* __restrict__ feat,
                              const float* __restrict__ xn,
                              const float* __restrict__ tx1,
                              const float* __restrict__ W0,
                              const float* __restrict__ W1,
                              const float* __restrict__ cb,
                              const float* __restrict__ tinW,
                              const float* __restrict__ tinb,
                              float* __restrict__ h) {
  int idx = blockIdx.x * blockDim.x + threadIdx.x;   // n*T + t
  if (idx >= NSEQ * GT) return;
  int n = idx / GT, t = idx % GT;
  int b = n / GC,  c = n % GC;

  float x0[GIN], x1[GIN];
  #pragma unroll
  for (int i = 0; i < GIN; i++) { x0[i] = xn[(long)idx * GIN + i]; x1[i] = tx1[(long)idx * GIN + i]; }

  float xf[2 * GIN];  // 28 inputs: [xn2(14), g(14)]
  xf[0] = pm25[((long)b * GT + t) * GC + c];
  const float* f2 = feat + (((long)b * (2 * GT) + (GT + t)) * GC + c) * (GIN - 1);
  #pragma unroll
  for (int i = 0; i < GIN - 1; i++) xf[1 + i] = f2[i];

  #pragma unroll
  for (int o = 0; o < GIN; o++) {     // GCN_OUT == 14 == GIN
    float s = cb[o];
    #pragma unroll
    for (int i = 0; i < GIN; i++) s += x0[i] * W0[i * GIN + o] + x1[i] * W1[i * GIN + o];
    xf[GIN + o] = 1.0f / (1.0f + __expf(-s));
  }

  float* out = h + (long)idx * GHID;
  const float ln1e4 = 9.210340371976184f;  // ln(10000)
  #pragma unroll
  for (int o = 0; o < GHID; o++) {
    float s = tinb[o];
    #pragma unroll
    for (int i = 0; i < 2 * GIN; i++) s += xf[i] * tinW[i * GHID + o];
    float freq = __expf(-ln1e4 * (float)(o & ~1) / (float)GHID);
    float ang  = (float)t * freq;
    s += (o & 1) ? __cosf(ang) : __sinf(ang);
    out[o] = s;
  }
}

// ---------------------------------------------------------------------------
// WMMA GEMM, compile-time K/N:  Y[M,N] = X[M,K] @ W[K,N] + bias (+relu)(+res)
// 128 threads = 4 wave32; block owns 64 rows (M % 64 == 0, grid exact).
// A tile and W (transposed [n][k] f16) staged in LDS via contiguous coalesced
// copies. Residual tile DMA'd LDS-ward asynchronously, overlapping WMMA.
// Epilogue routed through an LDS f32 tile -> fully coalesced global stores.
// ---------------------------------------------------------------------------
template <int K, int N, bool RELU, bool HASRES>
__global__ __launch_bounds__(128) void wmma_gemm(
    const float* __restrict__ X, const float* __restrict__ W,
    const float* __restrict__ bias, const float* __restrict__ resid,
    float* __restrict__ Y) {
  constexpr int KC   = (K + 31) / 32;
  constexpr int NT   = (N + 15) / 16;
  constexpr int KPAD = KC * 32;
  constexpr int NLD  = NT * 16;
  __shared__ __align__(16) _Float16 Wlds[NLD * KPAD];        // [n][k]
  __shared__ __align__(16) _Float16 Ast[64 * KPAD];          // [row][k]
  __shared__ __align__(16) float    Ylds[64 * N];            // output tile
  __shared__ __align__(16) float    Rlds[HASRES ? 64 * N : 4];

  const int tid = threadIdx.x;
  const long ybase = (long)blockIdx.x * 64 * N;

  if (HASRES) {
    // Kick off the async residual DMA first; it completes under the WMMAs.
    for (int idx = tid * 4; idx < 64 * N; idx += 128 * 4)
      async_g2l_b128(&Rlds[idx], &resid[ybase + idx]);
  }

  // 1) zero pad regions (KPAD>K, NLD>=N here)
  for (int idx = tid; idx < NLD * KPAD; idx += 128) Wlds[idx] = (_Float16)0.0f;
  for (int idx = tid; idx < 64 * KPAD; idx += 128) Ast[idx] = (_Float16)0.0f;
  __syncthreads();

  // 2) contiguous coalesced fills (f32 -> f16), scattered LDS stores
  for (int idx = tid; idx < K * N; idx += 128) {
    int k = idx / N, n = idx % N;
    Wlds[n * KPAD + k] = (_Float16)W[idx];
  }
  const long xbase = (long)blockIdx.x * 64 * K;   // 64-row tile is contiguous
  for (int idx = tid; idx < 64 * K; idx += 128) {
    int r = idx / K, k = idx % K;
    Ast[r * KPAD + k] = (_Float16)X[xbase + idx];
  }
  __syncthreads();

  const int wave = tid >> 5;
  const int lane = tid & 31;
  const int hi   = (lane >= 16) ? 1 : 0;
  const int arow = wave * 16 + (lane & 15);       // A row within block tile

  v8f acc[NT];
  #pragma unroll
  for (int i = 0; i < NT; i++) acc[i] = v8f{};

  #pragma unroll
  for (int kc = 0; kc < KC; kc++) {
    // CDNA5 16-bit A frag: lanes0-15 -> K 0-7|16-23 ; lanes16-31 -> 8-15|24-31
    v8h alo = *(const v8h*)&Ast[arow * KPAD + kc * 32 + (hi ? 8  : 0)];
    v8h ahi = *(const v8h*)&Ast[arow * KPAD + kc * 32 + (hi ? 24 : 16)];
    v16h afrag;
    #pragma unroll
    for (int e = 0; e < 8; e++) { afrag[e] = alo[e]; afrag[e + 8] = ahi[e]; }

    #pragma unroll
    for (int nt = 0; nt < NT; nt++) {
      // B frag: lane col = lane&15; lanes0-15 K 0-7|16-23, lanes16-31 K 8-15|24-31.
      int ncol = nt * 16 + (lane & 15);
      int kb   = kc * 32 + (hi ? 8 : 0);
      v8h blo = *(const v8h*)&Wlds[ncol * KPAD + kb];
      v8h bhi = *(const v8h*)&Wlds[ncol * KPAD + kb + 16];
      v16h bfrag;
      #pragma unroll
      for (int e = 0; e < 8; e++) { bfrag[e] = blo[e]; bfrag[e + 8] = bhi[e]; }
      acc[nt] = __builtin_amdgcn_wmma_f32_16x16x32_f16(
          false, afrag, false, bfrag, (short)0, acc[nt], false, false);
    }
  }

  // 3) park C/D fragments in the LDS output tile
  //    (C/D layout: VGPR r holds row r + hi*8; col = lane&15)
  const int yrow = wave * 16 + (hi ? 8 : 0);
  #pragma unroll
  for (int nt = 0; nt < NT; nt++) {
    int ncol = nt * 16 + (lane & 15);
    if (ncol < N) {
      #pragma unroll
      for (int r = 0; r < 8; r++) Ylds[(yrow + r) * N + ncol] = acc[nt][r];
    }
  }
  if (HASRES) wait_asynccnt0();   // residual DMA done before anyone reads Rlds
  __syncthreads();

  // 4) contiguous coalesced output pass: bias / relu / residual fused
  for (int idx = tid; idx < 64 * N; idx += 128) {
    float y = Ylds[idx] + bias[idx % N];
    if (RELU)   y = fmaxf(y, 0.0f);
    if (HASRES) y += Rlds[idx];
    Y[ybase + idx] = y;
  }
}

// ---------------------------------------------------------------------------
// per-(seq, head, query-row) softmax attention, T=24, HD=18
// ---------------------------------------------------------------------------
__global__ void attn_kernel(const float* __restrict__ q,
                            const float* __restrict__ k,
                            const float* __restrict__ v,
                            float* __restrict__ o) {
  int idx = blockIdx.x * blockDim.x + threadIdx.x;   // n*HEADS*T + hd*T + t
  if (idx >= NSEQ * GHEADS * GT) return;
  int t  = idx % GT;
  int hd = (idx / GT) % GHEADS;
  int n  = idx / (GT * GHEADS);
  const float scale = rsqrtf((float)GHD);
  long base = (long)n * GT * GHID + hd * GHD;

  float qr[GHD];
  #pragma unroll
  for (int d = 0; d < GHD; d++) qr[d] = q[base + (long)t * GHID + d];

  float s[GT]; float mx = -1e30f;
  for (int j = 0; j < GT; j++) {
    float a = 0.0f;
    #pragma unroll
    for (int d = 0; d < GHD; d++) a += qr[d] * k[base + (long)j * GHID + d];
    a *= scale; s[j] = a; mx = fmaxf(mx, a);
  }
  float sum = 0.0f;
  for (int j = 0; j < GT; j++) { s[j] = __expf(s[j] - mx); sum += s[j]; }
  float inv = 1.0f / sum;
  #pragma unroll
  for (int d = 0; d < GHD; d++) {
    float a = 0.0f;
    for (int j = 0; j < GT; j++) a += s[j] * v[base + (long)j * GHID + d];
    o[base + (long)t * GHID + d] = a * inv;
  }
}

__global__ void ln_kernel(const float* __restrict__ x,
                          const float* __restrict__ g,
                          const float* __restrict__ b,
                          float* __restrict__ y) {
  int r = blockIdx.x * blockDim.x + threadIdx.x;
  if (r >= MROWS) return;
  float v[GHID]; float m = 0.0f;
  #pragma unroll
  for (int i = 0; i < GHID; i++) { v[i] = x[(long)r * GHID + i]; m += v[i]; }
  m *= (1.0f / GHID);
  float var = 0.0f;
  #pragma unroll
  for (int i = 0; i < GHID; i++) { float d = v[i] - m; var += d * d; }
  var *= (1.0f / GHID);
  float inv = rsqrtf(var + 1e-5f);
  #pragma unroll
  for (int i = 0; i < GHID; i++)
    y[(long)r * GHID + i] = (v[i] - m) * inv * g[i] + b[i];
}

// out[b,t,c] = h[n=b*C+c, t] @ fc_W + fc_b
__global__ void fc_kernel(const float* __restrict__ h,
                          const float* __restrict__ W,
                          const float* __restrict__ b,
                          float* __restrict__ out) {
  int idx = blockIdx.x * blockDim.x + threadIdx.x;   // n*T + t
  if (idx >= NSEQ * GT) return;
  int n = idx / GT, t = idx % GT;
  int bb = n / GC, c = n % GC;
  float acc = b[0];
  #pragma unroll
  for (int i = 0; i < GHID; i++) acc += h[(long)idx * GHID + i] * W[i];
  out[((long)bb * GT + t) * GC + c] = acc;
}

// ---------------------------------------------------------------------------
extern "C" void kernel_launch(void* const* d_in, const int* in_sizes, int n_in,
                              void* d_out, int out_size, void* d_ws, size_t ws_size,
                              hipStream_t stream) {
  const float* pm25  = (const float*)d_in[0];
  const float* feat  = (const float*)d_in[1];
  const int*   edge  = (const int*)d_in[2];
  const float* chW0  = (const float*)d_in[3];
  const float* chW1  = (const float*)d_in[4];
  const float* chb   = (const float*)d_in[5];
  const float* tinW  = (const float*)d_in[6];
  const float* tinb  = (const float*)d_in[7];
  const float* Wq    = (const float*)d_in[8];
  const float* bq    = (const float*)d_in[9];
  const float* Wk    = (const float*)d_in[10];
  const float* bk    = (const float*)d_in[11];
  const float* Wv    = (const float*)d_in[12];
  const float* bv    = (const float*)d_in[13];
  const float* Wo    = (const float*)d_in[14];
  const float* bo    = (const float*)d_in[15];
  const float* ln1g  = (const float*)d_in[16];
  const float* ln1b  = (const float*)d_in[17];
  const float* fW1   = (const float*)d_in[18];
  const float* fb1   = (const float*)d_in[19];
  const float* fW2   = (const float*)d_in[20];
  const float* fb2   = (const float*)d_in[21];
  const float* ln2g  = (const float*)d_in[22];
  const float* ln2b  = (const float*)d_in[23];
  const float* fcW   = (const float*)d_in[24];
  const float* fcb   = (const float*)d_in[25];
  (void)n_in; (void)out_size; (void)ws_size;

  const int EG = in_sizes[2] / 2;

  // workspace layout (floats)
  float* ws = (float*)d_ws;
  size_t off = 0;
  float* xn   = ws + off; off += (size_t)NSEQ * GT * GIN;
  float* tx1  = ws + off; off += (size_t)NSEQ * GT * GIN;
  float* deg  = ws + off; off += NSEQ;
  float* dinv = ws + off; off += NSEQ;
  float* h    = ws + off; off += (size_t)MROWS * GHID;
  float* t1   = ws + off; off += (size_t)MROWS * GHID;
  float* R    = ws + off; off += (size_t)MROWS * GFFN;   // q/k/v/o OR ffn buf
  float* q  = R;
  float* kk = R + (size_t)MROWS * GHID;
  float* vv = R + (size_t)2 * MROWS * GHID;
  float* oo = R + (size_t)3 * MROWS * GHID;

  // --- graph conv front-end ---
  zero_f32<<<2048, 256, 0, stream>>>(deg, (long)NSEQ);
  zero_f32<<<2048, 256, 0, stream>>>(tx1, (long)NSEQ * GT * GIN);
  build_xn<<<(NSEQ * GT + 255) / 256, 256, 0, stream>>>(pm25, feat, xn);
  deg_kernel<<<(GB * EG + 255) / 256, 256, 0, stream>>>(edge, EG, deg);
  dinv_kernel<<<(NSEQ + 255) / 256, 256, 0, stream>>>(deg, dinv);
  scatter_kernel<<<(GB * EG * GT + 255) / 256, 256, 0, stream>>>(edge, EG, dinv, xn, tx1);
  fuse_h_kernel<<<(NSEQ * GT + 255) / 256, 256, 0, stream>>>(
      pm25, feat, xn, tx1, chW0, chW1, chb, tinW, tinb, h);

  // --- transformer encoder layers ---
  const int gemm_blocks = MROWS / 64;   // 6144, exact
  for (int l = 0; l < 2; l++) {
    const int wo36 = l * GHID * GHID, vb = l * GHID;
    wmma_gemm<GHID, GHID, false, false><<<gemm_blocks, 128, 0, stream>>>(
        h, Wq + wo36, bq + vb, nullptr, q);
    wmma_gemm<GHID, GHID, false, false><<<gemm_blocks, 128, 0, stream>>>(
        h, Wk + wo36, bk + vb, nullptr, kk);
    wmma_gemm<GHID, GHID, false, false><<<gemm_blocks, 128, 0, stream>>>(
        h, Wv + wo36, bv + vb, nullptr, vv);
    attn_kernel<<<(NSEQ * GHEADS * GT + 255) / 256, 256, 0, stream>>>(q, kk, vv, oo);
    wmma_gemm<GHID, GHID, false, true><<<gemm_blocks, 128, 0, stream>>>(
        oo, Wo + wo36, bo + vb, /*resid=*/h, t1);
    ln_kernel<<<(MROWS + 255) / 256, 256, 0, stream>>>(t1, ln1g + vb, ln1b + vb, h);
    wmma_gemm<GHID, GFFN, true, false><<<gemm_blocks, 128, 0, stream>>>(
        h, fW1 + l * GHID * GFFN, fb1 + l * GFFN, nullptr, R);
    wmma_gemm<GFFN, GHID, false, true><<<gemm_blocks, 128, 0, stream>>>(
        R, fW2 + l * GFFN * GHID, fb2 + vb, /*resid=*/h, t1);
    ln_kernel<<<(MROWS + 255) / 256, 256, 0, stream>>>(t1, ln2g + vb, ln2b + vb, h);
  }

  fc_kernel<<<(NSEQ * GT + 255) / 256, 256, 0, stream>>>(h, fcW, fcb, (float*)d_out);
}
